// GCNNet_5016521802311
// MI455X (gfx1250) — matrix operand order repaired and checked by reading, not compiled, not run on hardware
//
#include <hip/hip_runtime.h>
#include <hip/hip_bf16.h>
#include <stdint.h>

typedef __attribute__((ext_vector_type(16))) __bf16 v16bf;
typedef __attribute__((ext_vector_type(8)))  float  v8f;

#define N_NODES  100000
#define N_EDGES  1600000
#define N_GRAPHS 512
#define F_IN     768
#define F_HID    256
#define F_OUT    2

// ---------------------------------------------------------------- init
__global__ void k_init(float* __restrict__ deg, float* __restrict__ agg1,
                       float* __restrict__ agg2, float* __restrict__ counts,
                       float* __restrict__ out) {
    int64_t idx = (int64_t)blockIdx.x * blockDim.x + threadIdx.x;
    if (idx < (int64_t)N_NODES * F_HID) agg1[idx] = 0.0f;
    if (idx < N_NODES)                  deg[idx]  = 1.0f;   // self-loop
    if (idx < N_NODES * F_OUT)          agg2[idx] = 0.0f;
    if (idx < N_GRAPHS)                 counts[idx] = 0.0f;
    if (idx < N_GRAPHS * F_OUT)         out[idx] = 0.0f;
}

// ---------------------------------------------------------------- degree
__global__ void k_deg(const int* __restrict__ dst, float* __restrict__ deg) {
    int e = blockIdx.x * blockDim.x + threadIdx.x;
    if (e < N_EDGES) atomicAdd(&deg[dst[e]], 1.0f);
}

__global__ void k_dinv(const float* __restrict__ deg, float* __restrict__ dinv) {
    int i = blockIdx.x * blockDim.x + threadIdx.x;
    if (i < N_NODES) dinv[i] = rsqrtf(deg[i]);
}

// ------------------------------------------------- W1 -> bf16, transposed [256][768]
__global__ void k_w1t(const float* __restrict__ W1, __bf16* __restrict__ w1t) {
    int idx = blockIdx.x * blockDim.x + threadIdx.x;
    if (idx < F_HID * F_IN) {
        int n = idx / F_IN, k = idx % F_IN;
        w1t[idx] = (__bf16)W1[k * F_HID + n];
    }
}

// ------------------------------------------------- GEMM1: h1 = X @ W1 (bf16 WMMA, f32 acc)
// One wave per 16-row x 64-col output strip. grid = (6250, 4), block = 32 (wave32).
// A layout (16x32 bf16, as loaded here): lane<16 -> K[0..15]; lane>=16 -> K[16..31], M = lane&15
// B layout (32x16 bf16): lane<16 -> col n=lane, K[0..15]; lane>=16 -> col n=lane-16, K[16..31]
// C/D: VGPR r -> row r (lanes 0-15) / row r+8 (lanes 16-31), col = lane&15
__global__ void k_gemm1(const float* __restrict__ x, const __bf16* __restrict__ w1t,
                        float* __restrict__ h1) {
    const int lane = threadIdx.x;
    const int mt   = blockIdx.x;          // 0..6249  (100000 = 6250*16, no tail)
    const int n0   = blockIdx.y * 64;     // 0,64,128,192
    const int mrow = mt * 16 + (lane & 15);
    const int aoff = (lane & 16) ? 16 : 0;
    const int koff = (lane & 16) ? 16 : 0;
    const int ncol = lane & 15;

    v8f acc0 = {}, acc1 = {}, acc2 = {}, acc3 = {};

    const float*  xrow = x + (size_t)mrow * F_IN;
    const __bf16* wp0 = w1t + (size_t)(n0 +  0 + ncol) * F_IN + koff;
    const __bf16* wp1 = w1t + (size_t)(n0 + 16 + ncol) * F_IN + koff;
    const __bf16* wp2 = w1t + (size_t)(n0 + 32 + ncol) * F_IN + koff;
    const __bf16* wp3 = w1t + (size_t)(n0 + 48 + ncol) * F_IN + koff;

    for (int k0 = 0; k0 < F_IN; k0 += 32) {
        const float* xp = xrow + k0 + aoff;         // 32B-aligned
        __builtin_prefetch(xp + 32, 0, 0);          // next A tile -> global_prefetch_b8
        v16bf a;
#pragma unroll
        for (int j = 0; j < 8; ++j) a[j]     = (__bf16)xp[j];        // K = k0+aoff   .. +7
#pragma unroll
        for (int j = 0; j < 8; ++j) a[j + 8] = (__bf16)xp[j + 8];    // K = k0+aoff+8 .. +15

        v16bf b0 = *(const v16bf*)(wp0 + k0);
        v16bf b1 = *(const v16bf*)(wp1 + k0);
        v16bf b2 = *(const v16bf*)(wp2 + k0);
        v16bf b3 = *(const v16bf*)(wp3 + k0);

        acc0 = __builtin_amdgcn_wmma_f32_16x16x32_bf16(false, a, false, b0, (short)0, acc0, false, false);
        acc1 = __builtin_amdgcn_wmma_f32_16x16x32_bf16(false, a, false, b1, (short)0, acc1, false, false);
        acc2 = __builtin_amdgcn_wmma_f32_16x16x32_bf16(false, a, false, b2, (short)0, acc2, false, false);
        acc3 = __builtin_amdgcn_wmma_f32_16x16x32_bf16(false, a, false, b3, (short)0, acc3, false, false);
    }

    const int ms = mt * 16 + ((lane & 16) ? 8 : 0);
    float* outp = h1 + (size_t)ms * F_HID + n0 + ncol;
#pragma unroll
    for (int r = 0; r < 8; ++r) {
        outp[(size_t)r * F_HID +  0] = acc0[r];
        outp[(size_t)r * F_HID + 16] = acc1[r];
        outp[(size_t)r * F_HID + 32] = acc2[r];
        outp[(size_t)r * F_HID + 48] = acc3[r];
    }
}

// ---------------------------------------------- edge scatter, layer 1 (256 features)
// block = 256 threads (one per feature), grid = E. agg1 (102 MB) is L2-resident.
__global__ void k_scatter1(const float* __restrict__ h1, const int* __restrict__ src,
                           const int* __restrict__ dst, const float* __restrict__ dinv,
                           float* __restrict__ agg1) {
    int e = blockIdx.x;
    int f = threadIdx.x;
    int s = src[e], d = dst[e];
    float ne = dinv[s] * dinv[d];
    atomicAdd(&agg1[(size_t)d * F_HID + f], h1[(size_t)s * F_HID + f] * ne);
}

// --------------------------------- self-loop (1/deg = dinv^2) + bias + ReLU, in place
__global__ void k_selfrelu(const float* __restrict__ h1, const float* __restrict__ dinv,
                           const float* __restrict__ b1, float* __restrict__ agg1) {
    int64_t idx = (int64_t)blockIdx.x * blockDim.x + threadIdx.x;
    if (idx >= (int64_t)N_NODES * F_HID) return;
    int i = (int)(idx / F_HID);
    int f = (int)(idx % F_HID);
    float di = dinv[i];
    float v = agg1[idx] + h1[idx] * di * di + b1[f];
    agg1[idx] = fmaxf(v, 0.0f);
}

// --------------------------------------------- layer 2 GEMV: h2 = H @ W2 (256 -> 2)
__global__ void k_layer2(const float* __restrict__ H, const float* __restrict__ W2,
                         float* __restrict__ h2) {
    int i = blockIdx.x * blockDim.x + threadIdx.x;
    if (i >= N_NODES) return;
    const float* hp = H + (size_t)i * F_HID;
    float a0 = 0.0f, a1 = 0.0f;
#pragma unroll 8
    for (int f = 0; f < F_HID; ++f) {
        float v = hp[f];
        a0 += v * W2[2 * f + 0];
        a1 += v * W2[2 * f + 1];
    }
    h2[2 * i + 0] = a0;
    h2[2 * i + 1] = a1;
}

// ---------------------------------------------- edge scatter, layer 2 (2 features)
__global__ void k_scatter2(const float* __restrict__ h2, const int* __restrict__ src,
                           const int* __restrict__ dst, const float* __restrict__ dinv,
                           float* __restrict__ agg2) {
    int e = blockIdx.x * blockDim.x + threadIdx.x;
    if (e >= N_EDGES) return;
    int s = src[e], d = dst[e];
    float ne = dinv[s] * dinv[d];
    atomicAdd(&agg2[2 * d + 0], h2[2 * s + 0] * ne);
    atomicAdd(&agg2[2 * d + 1], h2[2 * s + 1] * ne);
}

// -------------------------- self loop + bias, then mean-pool sums/counts via atomics
__global__ void k_final(const float* __restrict__ h2, const float* __restrict__ agg2,
                        const float* __restrict__ dinv, const int* __restrict__ batch,
                        const float* __restrict__ b2, float* __restrict__ out_sums,
                        float* __restrict__ counts) {
    int i = blockIdx.x * blockDim.x + threadIdx.x;
    if (i >= N_NODES) return;
    float di = dinv[i];
    float sl = di * di;
    int g = batch[i];
    float v0 = agg2[2 * i + 0] + h2[2 * i + 0] * sl + b2[0];
    float v1 = agg2[2 * i + 1] + h2[2 * i + 1] * sl + b2[1];
    atomicAdd(&out_sums[2 * g + 0], v0);
    atomicAdd(&out_sums[2 * g + 1], v1);
    atomicAdd(&counts[g], 1.0f);
}

__global__ void k_pooldiv(float* __restrict__ out, const float* __restrict__ counts) {
    int g = blockIdx.x * blockDim.x + threadIdx.x;
    if (g >= N_GRAPHS) return;
    float c = fmaxf(counts[g], 1.0f);
    out[2 * g + 0] /= c;
    out[2 * g + 1] /= c;
}

// ---------------------------------------------------------------- launch
extern "C" void kernel_launch(void* const* d_in, const int* in_sizes, int n_in,
                              void* d_out, int out_size, void* d_ws, size_t ws_size,
                              hipStream_t stream) {
    const float* x     = (const float*)d_in[0];
    const int*   ei    = (const int*)d_in[1];
    const int*   batch = (const int*)d_in[2];
    const float* W1    = (const float*)d_in[3];
    const float* b1    = (const float*)d_in[4];
    const float* W2    = (const float*)d_in[5];
    const float* b2    = (const float*)d_in[6];
    const int*   src   = ei;             // edge_index[0]
    const int*   dst   = ei + N_EDGES;   // edge_index[1]
    float*       out   = (float*)d_out;

    // workspace carving (256B aligned; hipMalloc base is >=256B aligned)
    char* ws = (char*)d_ws;
    auto carve = [&](size_t bytes) -> char* {
        char* p = ws;
        ws += (bytes + 255) & ~(size_t)255;
        return p;
    };
    __bf16* w1t    = (__bf16*)carve((size_t)F_HID * F_IN * sizeof(__bf16));
    float*  h1     = (float*) carve((size_t)N_NODES * F_HID * sizeof(float));
    float*  agg1   = (float*) carve((size_t)N_NODES * F_HID * sizeof(float));
    float*  deg    = (float*) carve((size_t)N_NODES * sizeof(float));
    float*  dinv   = (float*) carve((size_t)N_NODES * sizeof(float));
    float*  h2     = (float*) carve((size_t)N_NODES * F_OUT * sizeof(float));
    float*  agg2   = (float*) carve((size_t)N_NODES * F_OUT * sizeof(float));
    float*  counts = (float*) carve((size_t)N_GRAPHS * sizeof(float));

    const int64_t nh = (int64_t)N_NODES * F_HID;

    k_init<<<(unsigned)((nh + 255) / 256), 256, 0, stream>>>(deg, agg1, agg2, counts, out);
    k_deg<<<(N_EDGES + 255) / 256, 256, 0, stream>>>(dst, deg);
    k_dinv<<<(N_NODES + 255) / 256, 256, 0, stream>>>(deg, dinv);
    k_w1t<<<(F_HID * F_IN + 255) / 256, 256, 0, stream>>>(W1, w1t);

    dim3 ggrid(N_NODES / 16, 4);            // 6250 x 4, exact (100000 % 16 == 0)
    k_gemm1<<<ggrid, 32, 0, stream>>>(x, w1t, h1);

    k_scatter1<<<N_EDGES, F_HID, 0, stream>>>(h1, src, dst, dinv, agg1);
    k_selfrelu<<<(unsigned)((nh + 255) / 256), 256, 0, stream>>>(h1, dinv, b1, agg1);
    k_layer2<<<(N_NODES + 255) / 256, 256, 0, stream>>>(agg1, W2, h2);
    k_scatter2<<<(N_EDGES + 255) / 256, 256, 0, stream>>>(h2, src, dst, dinv, agg2);
    k_final<<<(N_NODES + 255) / 256, 256, 0, stream>>>(h2, agg2, dinv, batch, b2, out, counts);
    k_pooldiv<<<(N_GRAPHS + 255) / 256, 256, 0, stream>>>(out, counts);
}
